// Block_8907762172244
// MI455X (gfx1250) — matrix operand order repaired and checked by reading, not compiled, not run on hardware
//
#include <hip/hip_runtime.h>
#include <hip/hip_bf16.h>

typedef __attribute__((ext_vector_type(16))) __bf16 v16bf;
typedef __attribute__((ext_vector_type(8)))  float  v8f;
typedef unsigned int u32;

#define CDIM 96
#define HDIM 64
#define WDIM 64
#define BDIM 16
#define NPIX (BDIM*HDIM*WDIM)   // 65536
#define HID2 384                // 2*HID
#define HID1 192
#define KTOT 2304               // 24*96

__device__ __forceinline__ float gelu_exact(float x) {
    return 0.5f * x * (1.0f + erff(x * 0.7071067811865476f));
}

// async copy of 16B (8 bf16) from global (saddr + voff) into LDS byte addr
__device__ __forceinline__ void async_b128(u32 ldsByteAddr, u32 globByteOff,
                                           const void* base) {
    asm volatile("global_load_async_to_lds_b128 %0, %1, %2"
                 :: "v"(ldsByteAddr), "v"(globByteOff), "s"(base)
                 : "memory");
}
__device__ __forceinline__ void async_wait0() {
    asm volatile("s_wait_asynccnt 0x0" ::: "memory");
}
__device__ __forceinline__ u32 lds_addr(const void* p) {
    return (u32)(size_t)p;   // LDS aperture: low 32 bits = LDS byte offset
}

// ---- WMMA fragment loaders (ISA 7.12.2 layouts, wave32) -------------------
// A (16x32 bf16): lane r (0..15) holds M=r; VGPR v<4 -> K=2v,2v+1 ; v>=4 -> K=16+2(v-4)
// lanes 16..31 same M, K offset +8 (and +24 for upper VGPRs).
__device__ __forceinline__ v16bf frag_a(const __bf16* lds, int row0, int kbase,
                                        int stride, int lane) {
    int half = (lane >> 4) & 1;
    int r    = lane & 15;
    union { v16bf v; u32 u[8]; } f;
    const __bf16* base = lds + (row0 + r) * stride + kbase + (half ? 8 : 0);
#pragma unroll
    for (int v = 0; v < 8; ++v) {
        int k = (v < 4) ? (2 * v) : (16 + 2 * (v - 4));
        f.u[v] = *(const u32*)(base + k);
    }
    return f.v;
}

// B (32x16 bf16), LDS stored N-major [n][k]: lane r (0..15) = N=r;
// VGPR v<4 -> K=2v,2v+1 ; v>=4 -> K=16+2(v-4) ; lanes 16..31 -> K offset +8.
__device__ __forceinline__ v16bf frag_b(const __bf16* lds, int n0, int kbase,
                                        int stride, int lane) {
    int half = (lane >> 4) & 1;
    int r    = lane & 15;
    union { v16bf v; u32 u[8]; } f;
    const __bf16* base = lds + (n0 + r) * stride + kbase + (half ? 8 : 0);
#pragma unroll
    for (int v = 0; v < 8; ++v) f.u[v] = *(const u32*)(base + ((v < 4) ? (2 * v) : (16 + 2 * (v - 4))));
    return f.v;
}

__device__ __forceinline__ v8f v8f_zero() {
    v8f z = {0.f, 0.f, 0.f, 0.f, 0.f, 0.f, 0.f, 0.f};
    return z;
}

// ===========================================================================
// K0: convert weights to bf16 (w2 [96][2304], pin_w [384][96], pout_w [96][192])
// ===========================================================================
__global__ void k_convert(const float* __restrict__ w2,
                          const float* __restrict__ pinw,
                          const float* __restrict__ poutw,
                          __bf16* __restrict__ w2bf,
                          __bf16* __restrict__ pinbf,
                          __bf16* __restrict__ poutbf) {
    int i = blockIdx.x * 256 + threadIdx.x;
    const int n1 = CDIM * KTOT;          // 221184
    const int n2 = HID2 * CDIM;          // 36864
    const int n3 = CDIM * HID1;          // 18432
    if (i < n1)               w2bf[i]           = (__bf16)w2[i];
    else if (i < n1 + n2)     pinbf[i - n1]     = (__bf16)pinw[i - n1];
    else if (i < n1 + n2 + n3) poutbf[i - n1 - n2] = (__bf16)poutw[i - n1 - n2];
}

// ===========================================================================
// K1: x += dwconv3x3(x); y = NHWC(x); y += LN3(y); z = LN1(y)
// one block per pixel, 96 threads (one per channel)
// ===========================================================================
__global__ void k_pre(const float* __restrict__ x,
                      const float* __restrict__ pos_w, const float* __restrict__ pos_b,
                      const float* __restrict__ ln3g, const float* __restrict__ ln3b,
                      const float* __restrict__ ln1g, const float* __restrict__ ln1b,
                      float* __restrict__ yBuf, float* __restrict__ zBuf) {
    __shared__ float sm[CDIM];
    int pixel = blockIdx.x;
    int b = pixel >> 12, h = (pixel >> 6) & 63, w = pixel & 63;
    int c = threadIdx.x;
    const float* xc = x + ((size_t)(b * CDIM + c)) * HDIM * WDIM;
    float s = pos_b[c];
#pragma unroll
    for (int p = 0; p < 3; ++p)
#pragma unroll
        for (int q = 0; q < 3; ++q) {
            int hh = h + p - 1, ww = w + q - 1;
            if (hh >= 0 && hh < HDIM && ww >= 0 && ww < WDIM)
                s += xc[hh * WDIM + ww] * pos_w[c * 9 + p * 3 + q];
        }
    float v = xc[h * WDIM + w] + s;

    sm[c] = v; __syncthreads();
    float m = 0.f, m2 = 0.f;
    for (int k = 0; k < CDIM; ++k) { float t = sm[k]; m += t; m2 += t * t; }
    m *= (1.0f / CDIM);
    float var = m2 * (1.0f / CDIM) - m * m;
    float y = v + (v - m) * rsqrtf(var + 1e-6f) * ln3g[c] + ln3b[c];

    __syncthreads(); sm[c] = y; __syncthreads();
    m = 0.f; m2 = 0.f;
    for (int k = 0; k < CDIM; ++k) { float t = sm[k]; m += t; m2 += t * t; }
    m *= (1.0f / CDIM);
    var = m2 * (1.0f / CDIM) - m * m;
    float z = (y - m) * rsqrtf(var + 1e-6f) * ln1g[c] + ln1b[c];

    yBuf[(size_t)pixel * CDIM + c] = y;
    zBuf[(size_t)pixel * CDIM + c] = z;
}

// ===========================================================================
// K2: fused patch-embed GEMM. Per block: 64 pixels, N=96, K=2304 in 24 chunks.
// cat features (1D convs along C + BN + GELU) generated on-the-fly into LDS,
// w2 chunks streamed global->LDS with GLOBAL_LOAD_ASYNC_TO_LDS_B128,
// WMMA bf16; epilogue = BN2+GELU + z residual + y residual.
// 256 threads = 8 waves; wave -> (m_tile = w>>1, 3 n-tiles).
// ===========================================================================
__global__ void k_patch(const float* __restrict__ zBuf, float* __restrict__ yBuf,
                        const __bf16* __restrict__ w2bf,
                        const float* __restrict__ w3a, const float* __restrict__ b3a,
                        const float* __restrict__ g3a, const float* __restrict__ bb3a,
                        const float* __restrict__ w3b, const float* __restrict__ b3b,
                        const float* __restrict__ g3b, const float* __restrict__ bb3b,
                        const float* __restrict__ w3c, const float* __restrict__ b3c,
                        const float* __restrict__ g3c, const float* __restrict__ bb3c,
                        const float* __restrict__ b2,
                        const float* __restrict__ bn2g, const float* __restrict__ bn2b) {
    __shared__ __attribute__((aligned(16))) float  zT[64 * CDIM];   // 24.0 KB
    __shared__ __attribute__((aligned(16))) __bf16 aT[64 * CDIM];   // 12.0 KB
    __shared__ __attribute__((aligned(16))) __bf16 bT[CDIM * CDIM]; // 18.4 KB

    const int tid  = threadIdx.x;
    const int lane = tid & 31;
    const int wave = tid >> 5;
    const int mtile = wave >> 1;            // 0..3
    const int ngrp  = wave & 1;             // 0..1 -> n tiles 3*ngrp..
    const int pix0  = blockIdx.x * 64;
    const float bnrs = rsqrtf(1.0f + 1e-5f);
    const u32 ldsbT = lds_addr(bT);

    for (int e = tid; e < 64 * CDIM; e += 256)
        zT[e] = zBuf[(size_t)pix0 * CDIM + e];
    __syncthreads();

    v8f acc[3] = {v8f_zero(), v8f_zero(), v8f_zero()};

    for (int o = 0; o < 24; ++o) {
        // w2 chunk -> bT, async global->LDS (row n: 96 bf16 = 12 b128 ops)
        for (int op = tid; op < CDIM * 12; op += 256) {
            int n  = op / 12;
            int kk = (op - n * 12) * 8;
            u32 lds  = ldsbT + (u32)(n * CDIM + kk) * 2;
            u32 voff = (u32)(n * KTOT + o * CDIM + kk) * 2;
            async_b128(lds, voff, w2bf);
        }

        int branch = o >> 3, oc = o & 7;
        const float* wv; float bias, bng, bnb; int ksz;
        if (branch == 0)      { wv = w3a + oc * 3; bias = b3a[oc]; bng = g3a[oc]; bnb = bb3a[oc]; ksz = 3; }
        else if (branch == 1) { wv = w3b + oc * 5; bias = b3b[oc]; bng = g3b[oc]; bnb = bb3b[oc]; ksz = 5; }
        else                  { wv = w3c + oc * 7; bias = b3c[oc]; bng = g3c[oc]; bnb = bb3c[oc]; ksz = 7; }
        float bnsc = bng * bnrs;
        int kh = ksz >> 1;

        // features -> aT (VALU work overlaps the async DMA)
        for (int e = tid; e < 64 * CDIM; e += 256) {
            int p = e / CDIM, c = e - p * CDIM;
            float sum = bias;
            for (int t = 0; t < ksz; ++t) {
                int cc = c + t - kh;
                if (cc >= 0 && cc < CDIM) sum += zT[p * CDIM + cc] * wv[t];
            }
            aT[e] = (__bf16)gelu_exact(sum * bnsc + bnb);
        }
        if (o + 1 < 24)
            __builtin_prefetch(w2bf + (size_t)(o + 1) * CDIM, 0, 1);
        async_wait0();
        __syncthreads();

#pragma unroll
        for (int ks = 0; ks < 3; ++ks) {
            v16bf af = frag_a(aT, mtile * 16, ks * 32, CDIM, lane);
#pragma unroll
            for (int j = 0; j < 3; ++j) {
                v16bf bf = frag_b(bT, (3 * ngrp + j) * 16, ks * 32, CDIM, lane);
                acc[j] = __builtin_amdgcn_wmma_f32_16x16x32_bf16(
                    false, af, false, bf, (short)0, acc[j], false, false);
            }
        }
        __syncthreads();
    }

    // epilogue: y += gelu((acc+b2)*bn2s + bn2b) + z
    int half = (lane >> 4) & 1, r = lane & 15;
#pragma unroll
    for (int j = 0; j < 3; ++j) {
        int c = (3 * ngrp + j) * 16 + r;
        float s2 = bn2g[c] * bnrs, sb = bn2b[c], bias2 = b2[c];
#pragma unroll
        for (int vr = 0; vr < 8; ++vr) {
            int p = mtile * 16 + vr + half * 8;
            size_t idx = (size_t)(pix0 + p) * CDIM + c;
            float val = gelu_exact((acc[j][vr] + bias2) * s2 + sb);
            yBuf[idx] = yBuf[idx] + val + zT[p * CDIM + c];
        }
    }
}

// ===========================================================================
// K3: zBuf = LN2(yBuf) per pixel
// ===========================================================================
__global__ void k_ln2(const float* __restrict__ yBuf,
                      const float* __restrict__ ln2g, const float* __restrict__ ln2b,
                      float* __restrict__ zBuf) {
    __shared__ float sm[CDIM];
    int pixel = blockIdx.x, c = threadIdx.x;
    float v = yBuf[(size_t)pixel * CDIM + c];
    sm[c] = v; __syncthreads();
    float m = 0.f, m2 = 0.f;
    for (int k = 0; k < CDIM; ++k) { float t = sm[k]; m += t; m2 += t * t; }
    m *= (1.0f / CDIM);
    float var = m2 * (1.0f / CDIM) - m * m;
    zBuf[(size_t)pixel * CDIM + c] = (v - m) * rsqrtf(var + 1e-6f) * ln2g[c] + ln2b[c];
}

// ===========================================================================
// K4: fully fused GDFN. One block = 8x8 spatial tile (+halo), 256 threads.
// pin WMMA (M=112 pad, N=384, K=96) -> u in LDS; 3x3 dwconv + gelu-gate in
// LDS; pout WMMA (M=64, K=192, N=96); epilogue *2 + transpose to d_out.
// Weights streamed global->LDS with async b128 DMA.
// dynamic LDS: a1 43008 | b1 73728 | u 86016 | g 24576 | b2w 36864 = 264192 B
// ===========================================================================
__global__ void k_ffn(const float* __restrict__ yn,          // zBuf (post-LN2, NHWC)
                      const __bf16* __restrict__ pinbf,      // [384][96]
                      const float* __restrict__ pin_b,
                      const float* __restrict__ dw_w, const float* __restrict__ dw_b,
                      const __bf16* __restrict__ poutbf,     // [96][192]
                      const float* __restrict__ pout_b,
                      float* __restrict__ out) {
    extern __shared__ __attribute__((aligned(16))) char smem[];
    __bf16* a1  = (__bf16*)(smem);                 // [112][96]
    __bf16* b1  = (__bf16*)(smem + 43008);         // [384][96]
    __bf16* uT  = (__bf16*)(smem + 116736);        // [112][384]
    __bf16* gT  = (__bf16*)(smem + 202752);        // [64][192]
    __bf16* b2w = (__bf16*)(smem + 227328);        // [96][192]

    const int tid  = threadIdx.x;
    const int lane = tid & 31;
    const int wave = tid >> 5;
    const int bt = blockIdx.x;
    const int b  = bt >> 6;
    const int tt = bt & 63;
    const int x0 = (tt >> 3) * 8;   // x == W coordinate
    const int y0 = (tt & 7) * 8;    // y == H coordinate

    // ---- stage A: async-DMA both weight matrices into LDS, then halo loads
    {
        u32 ldsb1 = lds_addr(b1);
        for (int op = tid; op < (HID2 * CDIM) / 8; op += 256)   // 4608 ops
            async_b128(ldsb1 + (u32)op * 16, (u32)op * 16, pinbf);
        u32 ldsb2 = lds_addr(b2w);
        for (int op = tid; op < (CDIM * HID1) / 8; op += 256)   // 2304 ops
            async_b128(ldsb2 + (u32)op * 16, (u32)op * 16, poutbf);
    }
    for (int e = tid; e < 112 * CDIM; e += 256) {
        int m = e / CDIM, c = e - m * CDIM;
        float v = 0.f;
        if (m < 100) {
            int i = m / 10, j = m - i * 10;
            int gx = x0 - 1 + i, gy = y0 - 1 + j;
            if (gx >= 0 && gx < WDIM && gy >= 0 && gy < HDIM)
                v = yn[(((size_t)b * HDIM + gy) * WDIM + gx) * CDIM + c];
        }
        a1[e] = (__bf16)v;
    }
    __builtin_prefetch(dw_w, 0, 2);
    async_wait0();
    __syncthreads();

    // ---- stage B: u = pin(yn) + bias via WMMA; 7 m-tiles x 24 n-tiles
    for (int t = wave; t < 7 * 24; t += 8) {
        int mt = t / 24, nt = t - mt * 24;
        v8f acc = v8f_zero();
#pragma unroll
        for (int ks = 0; ks < 3; ++ks) {
            v16bf af = frag_a(a1, mt * 16, ks * 32, CDIM, lane);
            v16bf bf = frag_b(b1, nt * 16, ks * 32, CDIM, lane);
            acc = __builtin_amdgcn_wmma_f32_16x16x32_bf16(
                false, af, false, bf, (short)0, acc, false, false);
        }
        int half = (lane >> 4) & 1, r = lane & 15;
        int oc = nt * 16 + r;
        float bias = pin_b[oc];
#pragma unroll
        for (int vr = 0; vr < 8; ++vr) {
            int m = mt * 16 + vr + half * 8;
            uT[m * HID2 + oc] = (__bf16)(acc[vr] + bias);
        }
    }
    __syncthreads();

    // ---- stage C: 3x3 depthwise conv + GELU gating (interior 64 px x 192 ch)
    for (int e = tid; e < 64 * HID1; e += 256) {
        int q = e / HID1, ch = e - q * HID1;
        int ix = q >> 3, iy = q & 7;          // interior coords 0..7
        float s1 = 0.f, s2 = 0.f;
#pragma unroll
        for (int p = 0; p < 3; ++p)
#pragma unroll
            for (int r2 = 0; r2 < 3; ++r2) {
                int gx = x0 + ix + p - 1, gy = y0 + iy + r2 - 1;
                if (gx >= 0 && gx < WDIM && gy >= 0 && gy < HDIM) {
                    int row = (ix + p) * 10 + (iy + r2);
                    s1 += (float)uT[row * HID2 + ch]        * dw_w[ch * 9 + p * 3 + r2];
                    s2 += (float)uT[row * HID2 + HID1 + ch] * dw_w[(HID1 + ch) * 9 + p * 3 + r2];
                }
            }
        float g1 = s1 + dw_b[ch];
        float g2 = s2 + dw_b[HID1 + ch];
        gT[e] = (__bf16)(gelu_exact(g1) * g2);
    }
    __syncthreads();

    // ---- stage D: pout WMMA (M=64, K=192, N=96) + *2 epilogue to d_out
    const int mtile = wave >> 1;
    const int ngrp  = wave & 1;
    v8f acc[3] = {v8f_zero(), v8f_zero(), v8f_zero()};
#pragma unroll
    for (int ks = 0; ks < 6; ++ks) {
        v16bf af = frag_a(gT, mtile * 16, ks * 32, HID1, lane);
#pragma unroll
        for (int j = 0; j < 3; ++j) {
            v16bf bf = frag_b(b2w, (3 * ngrp + j) * 16, ks * 32, HID1, lane);
            acc[j] = __builtin_amdgcn_wmma_f32_16x16x32_bf16(
                false, af, false, bf, (short)0, acc[j], false, false);
        }
    }
    int half = (lane >> 4) & 1, r = lane & 15;
#pragma unroll
    for (int j = 0; j < 3; ++j) {
        int co = (3 * ngrp + j) * 16 + r;
        float bias = pout_b[co];
#pragma unroll
        for (int vr = 0; vr < 8; ++vr) {
            int q = mtile * 16 + vr + half * 8;
            int gx = x0 + (q >> 3), gy = y0 + (q & 7);
            out[(((size_t)b * CDIM + co) * WDIM + gx) * HDIM + gy] =
                2.0f * (acc[j][vr] + bias);
        }
    }
}

// ===========================================================================
extern "C" void kernel_launch(void* const* d_in, const int* in_sizes, int n_in,
                              void* d_out, int out_size, void* d_ws, size_t ws_size,
                              hipStream_t stream) {
    const float* x      = (const float*)d_in[0];
    const float* pos_w  = (const float*)d_in[1];
    const float* pos_b  = (const float*)d_in[2];
    const float* ln1_g  = (const float*)d_in[3];
    const float* ln1_b  = (const float*)d_in[4];
    const float* ln2_g  = (const float*)d_in[5];
    const float* ln2_b  = (const float*)d_in[6];
    const float* ln3_g  = (const float*)d_in[7];
    const float* ln3_b  = (const float*)d_in[8];
    const float* w3a    = (const float*)d_in[9];
    const float* b3a    = (const float*)d_in[10];
    const float* g3a    = (const float*)d_in[11];
    const float* bb3a   = (const float*)d_in[12];
    const float* w3b    = (const float*)d_in[13];
    const float* b3b    = (const float*)d_in[14];
    const float* g3b    = (const float*)d_in[15];
    const float* bb3b   = (const float*)d_in[16];
    const float* w3c    = (const float*)d_in[17];
    const float* b3c    = (const float*)d_in[18];
    const float* g3c    = (const float*)d_in[19];
    const float* bb3c   = (const float*)d_in[20];
    const float* w2     = (const float*)d_in[21];
    const float* b2     = (const float*)d_in[22];
    const float* bn2_g  = (const float*)d_in[23];
    const float* bn2_b  = (const float*)d_in[24];
    const float* pin_w  = (const float*)d_in[25];
    const float* pin_b  = (const float*)d_in[26];
    const float* dw_w   = (const float*)d_in[27];
    const float* dw_b   = (const float*)d_in[28];
    const float* pout_w = (const float*)d_in[29];
    const float* pout_b = (const float*)d_in[30];

    char* ws = (char*)d_ws;
    float*  yBuf   = (float*)(ws);                                 // 25165824 B
    float*  zBuf   = (float*)(ws + 25165824);                      // 25165824 B
    __bf16* w2bf   = (__bf16*)(ws + 50331648);                     //   442368 B
    __bf16* pinbf  = (__bf16*)(ws + 50774016);                     //    73728 B
    __bf16* poutbf = (__bf16*)(ws + 50847744);                     //    36864 B

    // K0: weight conversion (276480 elements total)
    k_convert<<<(276480 + 255) / 256, 256, 0, stream>>>(
        w2, pin_w, pout_w, w2bf, pinbf, poutbf);

    // K1: pos dwconv + residual, LN3 residual, LN1
    k_pre<<<NPIX, CDIM, 0, stream>>>(x, pos_w, pos_b, ln3_g, ln3_b, ln1_g, ln1_b,
                                     yBuf, zBuf);

    // K2: fused patch-embed GEMM (M=65536, K=2304, N=96) via bf16 WMMA
    k_patch<<<NPIX / 64, 256, 0, stream>>>(
        zBuf, yBuf, w2bf,
        w3a, b3a, g3a, bb3a, w3b, b3b, g3b, bb3b, w3c, b3c, g3c, bb3c,
        b2, bn2_g, bn2_b);

    // K3: LN2
    k_ln2<<<NPIX, CDIM, 0, stream>>>(yBuf, ln2_g, ln2_b, zBuf);

    // K4: fused GDFN (pin WMMA -> dwconv+gate in LDS -> pout WMMA -> out)
    k_ffn<<<BDIM * 64, 256, 264192, stream>>>(
        zBuf, pinbf, pin_b, dw_w, dw_b, poutbf, pout_b, (float*)d_out);
}
